// HardAndLayer_9672266350669
// MI455X (gfx1250) — 8-bit matrix operand order repaired and checked by reading, compiled, not run on hardware
//
#include <hip/hip_runtime.h>
#include <hip/hip_bf16.h>
#include <stdint.h>

#define LAYER_SIZE 8192
#define INPUT_DIM  32768

#define KSPLIT          16
#define TILE_ROWS       16
#define NTILES          (LAYER_SIZE / TILE_ROWS)
#define K_PER_SPLIT     (INPUT_DIM / KSPLIT)
#define KITERS          (K_PER_SPLIT / 64)
#define WAVES_TOTAL     (NTILES * KSPLIT)
#define WAVES_PER_BLOCK 8
#define BLOCK_THREADS   (WAVES_PER_BLOCK * 32)

typedef __attribute__((ext_vector_type(8))) int v8i;
typedef __attribute__((ext_vector_type(4))) int v4i;

__global__ void prep_kernel(const uint32_t* __restrict__ x_dw,
                            uint32_t* __restrict__ notx_dw,
                            int* __restrict__ counts) {
    const int i = blockIdx.x * blockDim.x + threadIdx.x;
    counts[i] = 0;
    notx_dw[i] = 0x01010101u - x_dw[i];
}

__global__ __launch_bounds__(BLOCK_THREADS)
void hard_and_wmma_kernel(const uint8_t* __restrict__ w,
                          const uint8_t* __restrict__ notx,
                          int* __restrict__ counts) {
    const int wave = blockIdx.x * WAVES_PER_BLOCK + (int)(threadIdx.x >> 5);
    const int lane = threadIdx.x & 31;
    const int tile = wave >> 4;
    const int ks   = wave & (KSPLIT - 1);
    const int row0 = tile * TILE_ROWS;
    const int kbase = ks * K_PER_SPLIT;

    const int r  = lane & 15;
    const int hi = lane >> 4;
    const uint8_t* __restrict__ wrow = w + (size_t)(row0 + r) * INPUT_DIM;

    v8i acc = {};

    for (int it = 0; it < KITERS; ++it) {
        const int k = kbase + (it << 6);

        const int ka = k + hi * 16;
        union { unsigned long long q[4]; v8i v; } A;
        const unsigned long long* pa = (const unsigned long long*)(wrow + ka);
        A.q[0] = __builtin_nontemporal_load(pa + 0);
        A.q[1] = __builtin_nontemporal_load(pa + 1);
        A.q[2] = __builtin_nontemporal_load(pa + 4);
        A.q[3] = __builtin_nontemporal_load(pa + 5);

        const int kb = k + hi * 16;
        union { v4i h[2]; v8i v; } B;
        B.h[0] = *(const v4i*)(notx + kb);
        B.h[1] = *(const v4i*)(notx + kb + 32);

        acc = __builtin_amdgcn_wmma_i32_16x16x64_iu8(
             false, A.v,  false, B.v, acc,
             false,  false);
    }

    if (r == 0) {
        #pragma unroll
        for (int j = 0; j < 8; ++j) {
            atomicAdd(&counts[row0 + hi * 8 + j], acc[j]);
        }
    }
}

__global__ void finalize_kernel(const int* __restrict__ counts,
                                float* __restrict__ out) {
    const int i = blockIdx.x * blockDim.x + threadIdx.x;
    out[i] = (counts[i] == 0) ? 1.0f : 0.0f;
}

extern "C" void kernel_launch(void* const* d_in, const int* in_sizes, int n_in,
                              void* d_out, int out_size, void* d_ws, size_t ws_size,
                              hipStream_t stream) {
    const uint8_t* w = (const uint8_t*)d_in[0];
    const uint8_t* x = (const uint8_t*)d_in[1];
    float* out = (float*)d_out;

    int*     counts = (int*)d_ws;
    uint8_t* notx   = (uint8_t*)d_ws + LAYER_SIZE * sizeof(int);

    prep_kernel<<<LAYER_SIZE / 256, 256, 0, stream>>>(
        (const uint32_t*)x, (uint32_t*)notx, counts);
    hard_and_wmma_kernel<<<WAVES_TOTAL / WAVES_PER_BLOCK, BLOCK_THREADS, 0, stream>>>(
        w, notx, counts);
    finalize_kernel<<<LAYER_SIZE / 256, 256, 0, stream>>>(counts, out);
}
